// VmfOrdinalLossVectorized_43722767073364
// MI455X (gfx1250) — matrix operand order repaired and checked by reading, not compiled, not run on hardware
//
#include <hip/hip_runtime.h>
#include <hip/hip_bf16.h>
#include <math.h>

typedef __attribute__((ext_vector_type(16))) _Float16 v16h;
typedef __attribute__((ext_vector_type(8)))  _Float16 v8h;
typedef __attribute__((ext_vector_type(8)))  float    v8f;

#define KC 64
#define DD 128
#define LN_HALF (-0.6931471805599453f)
#define LOG_2PI 1.8378770664093453f

// ---------------------------------------------------------------- zero init
__global__ void vmf_zero(float* p, int n, float* out) {
    int i = blockIdx.x * blockDim.x + threadIdx.x;
    if (i < n) p[i] = 0.0f;
    if (i == 0) out[0] = 0.0f;
}

// ---------------------------------------------------------------- mus prep
// one thread per class k: normalize mu row, kappa = softplus, logC, A
__global__ void vmf_prep_mus(const float* __restrict__ mus,
                             const float* __restrict__ rho,
                             float* __restrict__ musN,
                             _Float16* __restrict__ musN_h,
                             float* __restrict__ kappas,
                             float* __restrict__ logC,
                             float* __restrict__ Avec) {
    int k = threadIdx.x;
    if (k >= KC) return;
    const float* row = mus + k * DD;
    float ss = 0.0f;
    for (int d = 0; d < DD; ++d) { float x = row[d]; ss += x * x; }
    float rinv = 1.0f / fmaxf(sqrtf(ss), 1e-12f);
    for (int d = 0; d < DD; ++d) {
        float v = row[d] * rinv;
        musN[k * DD + d]   = v;
        musN_h[k * DD + d] = (_Float16)v;
    }
    float x  = rho[k];
    float sp = (x > 20.0f) ? x : log1pf(expf(x));
    float kap = fmaxf(sp, 1e-6f);
    kappas[k] = kap;
    // logC_d(kappa, 128): v = 63, half_d = 64
    float logI;
    if (kap < 1e-3f)
        logI = 63.0f * logf(kap * 0.5f + 1e-12f) - lgammaf(64.0f);
    else
        logI = kap - 0.5f * logf(6.283185307179586f * kap + 1e-12f);
    logC[k] = -63.0f * logf(kap + 1e-12f) - 64.0f * LOG_2PI - logI;
    // approx A_d
    float kc = fmaxf(kap, 1e-8f);
    Avec[k] = (kc > 50.0f) ? (1.0f - 127.0f / (2.0f * kc)) : (kc / 128.0f);
}

// ---------------------------------------------------------------- features
// one wave32 per sample row: normalize, store f16, atomically accumulate
// per-class sums and counts
__global__ void vmf_norm_feat(const float* __restrict__ features,
                              const int* __restrict__ labels,
                              _Float16* __restrict__ featN_h,
                              float* __restrict__ sums,
                              float* __restrict__ counts, int N) {
    int wid  = threadIdx.x >> 5;
    int lane = threadIdx.x & 31;
    int n = blockIdx.x * 8 + wid;
    if (n >= N) return;
    float4 f = ((const float4*)(features + (size_t)n * DD))[lane];
    float ss = f.x * f.x + f.y * f.y + f.z * f.z + f.w * f.w;
    for (int o = 16; o > 0; o >>= 1) ss += __shfl_xor(ss, o, 32);
    float rinv = 1.0f / fmaxf(sqrtf(ss), 1e-12f);
    float vx = f.x * rinv, vy = f.y * rinv, vz = f.z * rinv, vw = f.w * rinv;
    _Float16* dst = featN_h + (size_t)n * DD + lane * 4;
    dst[0] = (_Float16)vx; dst[1] = (_Float16)vy;
    dst[2] = (_Float16)vz; dst[3] = (_Float16)vw;
    int c = labels[n];
    float* s = sums + c * DD + lane * 4;
    atomicAdd(s + 0, vx); atomicAdd(s + 1, vy);
    atomicAdd(s + 2, vz); atomicAdd(s + 3, vw);
    if (lane == 0) atomicAdd(counts + c, 1.0f);
}

// ---------------------------------------------------------------- WMMA GEMM
// L[n,c] = logC[c] + kappa[c] * dot(featN[n,:], musN[c,:])
// one wave per 16x16 output tile, K-loop of 4 x (16x16x32 f16 WMMA)
__global__ void vmf_gemm_L(const _Float16* __restrict__ featN_h,
                           const _Float16* __restrict__ musN_h,
                           const float* __restrict__ kappas,
                           const float* __restrict__ logC,
                           float* __restrict__ Lmat, int N) {
    int wid  = threadIdx.x >> 5;
    int lane = threadIdx.x & 31;
    int tile = blockIdx.x * 8 + wid;
    int ntiles = (N / 16) * (KC / 16);
    if (tile >= ntiles) return;
    int row0 = (tile >> 2) * 16;       // sample tile
    int c0   = (tile & 3) * 16;        // class tile
    int lmod = lane & 15, lhi = lane >> 4;

    // A operand: row = row0+lmod, halves 0..7 -> K = k0 + lhi*8 + i,
    //            halves 8..15 -> K = k0 + 16 + lhi*8 + i
    const _Float16* abase = featN_h + (size_t)(row0 + lmod) * DD + lhi * 8;
    // B operand: col = c0+lmod, halves 0..7 -> K = k0 + lhi*8 + i,
    //            halves 8..15 -> K = k0 + 16 + lhi*8 + i
    const _Float16* bbase = musN_h + (size_t)(c0 + lmod) * DD + lhi * 8;

    v8f acc = {};
    for (int kk = 0; kk < 4; ++kk) {
        int k0 = kk * 32;
        v8h a0 = *(const v8h*)(abase + k0);
        v8h a1 = *(const v8h*)(abase + k0 + 16);
        v8h b0 = *(const v8h*)(bbase + k0);
        v8h b1 = *(const v8h*)(bbase + k0 + 16);
        v16h a, b;
        for (int i = 0; i < 8; ++i) {
            a[i] = a0[i]; a[i + 8] = a1[i];
            b[i] = b0[i]; b[i + 8] = b1[i];
        }
        acc = __builtin_amdgcn_wmma_f32_16x16x32_f16(
            false, a, false, b, (short)0, acc, false, false);
    }
    int col = c0 + lmod;
    float kap = kappas[col], lc = logC[col];
    for (int v = 0; v < 8; ++v) {
        int n = row0 + v + lhi * 8;    // C/D layout: vgpr v, lanes16-31 -> M+8
        Lmat[(size_t)n * KC + col] = lc + kap * acc[v];
    }
}

// ---------------------------------------------------------------- JS accum
// one wave per sample: numer[c,j] += Lc - logaddexp(Lc+ln2h, Lj+ln2h)
__global__ void vmf_js_accum(const float* __restrict__ Lmat,
                             const int* __restrict__ labels,
                             float* __restrict__ numer, int N) {
    int wid  = threadIdx.x >> 5;
    int lane = threadIdx.x & 31;
    int n = blockIdx.x * 8 + wid;
    if (n >= N) return;
    int cl = labels[n];
    float Lc = Lmat[(size_t)n * KC + cl];
    for (int rep = 0; rep < 2; ++rep) {
        int j = lane + rep * 32;
        float Lj = Lmat[(size_t)n * KC + j];
        float d = fabsf(Lc - Lj);
        float logm = fmaxf(Lc, Lj) + log1pf(expf(-d)) + LN_HALF;
        atomicAdd(numer + cl * KC + j, Lc - logm);
    }
}

// ---------------------------------------------------------------- rank loss
// single block: means in LDS, means@musN^T, hinge reduce
__global__ void vmf_rank(const float* __restrict__ sums,
                         const float* __restrict__ counts,
                         const float* __restrict__ musN,
                         const float* __restrict__ kappas,
                         float* __restrict__ out) {
    __shared__ float means[KC * DD];   // 32 KB
    __shared__ float mdm[KC * KC];     // 16 KB
    __shared__ float red[8];
    int tid = threadIdx.x;
    for (int idx = tid; idx < KC * DD; idx += 256) {
        int k = idx >> 7;
        float cnt = counts[k];
        means[idx] = (cnt > 0.0f) ? sums[idx] / cnt : musN[idx];
    }
    __syncthreads();
    for (int idx = tid; idx < KC * KC; idx += 256) {
        int i = idx >> 6, j = idx & 63;
        float s = 0.0f;
        for (int d = 0; d < DD; ++d) s += means[i * DD + d] * musN[j * DD + d];
        mdm[idx] = s;
    }
    __syncthreads();
    float local = 0.0f;
    for (int idx = tid; idx < KC * KC; idx += 256) {
        int i = idx >> 6, j = idx & 63;
        if (i == j) continue;
        float E_si = kappas[i] * mdm[i * KC + i];
        float diff = E_si - kappas[j] * mdm[idx];
        float hinge = fmaxf(0.0f, 0.5f * fabsf((float)(i - j)) - diff);
        local += hinge / fmaxf(counts[i], 1.0f);
    }
    for (int o = 16; o > 0; o >>= 1) local += __shfl_down(local, o, 32);
    if ((tid & 31) == 0) red[tid >> 5] = local;
    __syncthreads();
    if (tid == 0) {
        float t = 0.0f;
        for (int i = 0; i < 8; ++i) t += red[i];
        atomicAdd(out, t / ((float)(KC * KC) + 1e-9f));   // LAMBDA_RANK = 1
    }
}

// ---------------------------------------------------------------- JS final
__global__ void vmf_js_final(const float* __restrict__ numer,
                             const float* __restrict__ counts,
                             const float* __restrict__ musN,
                             const float* __restrict__ kappas,
                             const float* __restrict__ logC,
                             const float* __restrict__ Avec,
                             float* __restrict__ out) {
    __shared__ float mdot[KC * KC];    // 16 KB
    __shared__ float redv[8], redw[8];
    int tid = threadIdx.x;
    for (int idx = tid; idx < KC * KC; idx += 256) {
        int i = idx >> 6, j = idx & 63;
        float s = 0.0f;
        for (int d = 0; d < DD; ++d) s += musN[i * DD + d] * musN[j * DD + d];
        mdot[idx] = s;
    }
    __syncthreads();
    float vs = 0.0f, ws = 0.0f;
    for (int idx = tid; idx < KC * KC; idx += 256) {
        int i = idx >> 6, j = idx & 63;
        if (i == j) continue;
        float w = fabsf((float)(i - j));
        float ci = counts[i], cj = counts[j];
        float js;
        if (ci == 0.0f || cj == 0.0f) {
            float dot  = mdot[idx];
            float klij = logC[i] - logC[j] + Avec[i] * (kappas[i] - kappas[j] * dot);
            float klji = logC[j] - logC[i] + Avec[j] * (kappas[j] - kappas[i] * dot);
            js = 0.5f * (klij + klji);
        } else {
            float kl_pi = numer[i * KC + j] / fmaxf(ci, 1.0f);
            float kl_pj = numer[j * KC + i] / fmaxf(cj, 1.0f);  // numer_pj = numer_pi^T
            js = 0.5f * (kl_pi + kl_pj);
        }
        vs += w * js;
        ws += w;
    }
    for (int o = 16; o > 0; o >>= 1) { vs += __shfl_down(vs, o, 32); ws += __shfl_down(ws, o, 32); }
    if ((tid & 31) == 0) { redv[tid >> 5] = vs; redw[tid >> 5] = ws; }
    __syncthreads();
    if (tid == 0) {
        float tv = 0.0f, tw = 0.0f;
        for (int i = 0; i < 8; ++i) { tv += redv[i]; tw += redw[i]; }
        atomicAdd(out, tv / (tw + 1e-9f));                // LAMBDA_JS = 1
    }
}

// ---------------------------------------------------------------- launch
extern "C" void kernel_launch(void* const* d_in, const int* in_sizes, int n_in,
                              void* d_out, int out_size, void* d_ws, size_t ws_size,
                              hipStream_t stream) {
    const float* features = (const float*)d_in[0];
    const float* mus      = (const float*)d_in[1];
    const float* rho      = (const float*)d_in[2];
    const int*   labels   = (const int*)d_in[3];
    float* out = (float*)d_out;
    const int N = in_sizes[0] / DD;

    // workspace layout (floats); zero-init region first
    float* w      = (float*)d_ws;
    float* counts = w;                       // 64
    float* sums   = counts + KC;             // 64*128
    float* numer  = sums + KC * DD;          // 64*64
    const int nzero = KC + KC * DD + KC * KC;
    float* musN   = numer + KC * KC;         // 64*128
    float* kappas = musN + KC * DD;          // 64
    float* logC   = kappas + KC;             // 64
    float* Avec   = logC + KC;               // 64
    float* Lmat   = Avec + KC;               // N*64
    _Float16* musN_h  = (_Float16*)(Lmat + (size_t)N * KC);     // 64*128 halves
    _Float16* featN_h = musN_h + KC * DD;                       // N*128 halves

    vmf_zero<<<(nzero + 255) / 256, 256, 0, stream>>>(counts, nzero, out);
    vmf_prep_mus<<<1, 64, 0, stream>>>(mus, rho, musN, musN_h, kappas, logC, Avec);
    vmf_norm_feat<<<(N + 7) / 8, 256, 0, stream>>>(features, labels, featN_h, sums, counts, N);
    {
        int ntiles = (N / 16) * (KC / 16);
        vmf_gemm_L<<<(ntiles + 7) / 8, 256, 0, stream>>>(featN_h, musN_h, kappas, logC, Lmat, N);
    }
    vmf_js_accum<<<(N + 7) / 8, 256, 0, stream>>>(Lmat, labels, numer, N);
    vmf_rank<<<1, 256, 0, stream>>>(sums, counts, musN, kappas, out);
    vmf_js_final<<<1, 256, 0, stream>>>(numer, counts, musN, kappas, logC, Avec, out);
}
